// IDCT2_torch_26645977104744
// MI455X (gfx1250) — matrix operand order repaired and checked by reading, not compiled, not run on hardware
//
#include <hip/hip_runtime.h>
#include <hip/hip_bf16.h>

// 2D IDCT (ortho): Y = A * X * A^T per 256x256 image (512 images).
// Fused per-image workgroup; bf16 hi/lo split WMMA (3 products) for
// ~fp32 accuracy at bf16 matrix rate; intermediate T transposed in LDS
// (pitch 257). Register-blocked 4-tile accumulation so each hi/lo split
// feeds 12 WMMAs; fragments packed dword-wise to avoid b16 move churn.

typedef __attribute__((ext_vector_type(16))) __bf16         v16bf;
typedef __attribute__((ext_vector_type(16))) unsigned short v16u;
typedef __attribute__((ext_vector_type(8)))  unsigned short v8u;
typedef __attribute__((ext_vector_type(8)))  unsigned int   v8i;
typedef __attribute__((ext_vector_type(8)))  float          v8f;

#define DN 256
#define LDS_PITCH 257

__device__ __forceinline__ unsigned short bf16_trunc(float f) {
  return (unsigned short)(__float_as_uint(f) >> 16);
}
__device__ __forceinline__ float bf16_to_f(unsigned short u) {
  return __uint_as_float(((unsigned int)u) << 16);
}

// IDCT-III (ortho) basis A[n,k] = w[k]*cos(pi*(2n+1)*k/(2N)) as bf16 hi/lo.
__global__ void build_basis(unsigned short* __restrict__ Ah,
                            unsigned short* __restrict__ Al) {
  const int n = blockIdx.x;
  const int k = threadIdx.x;
  const float PI = 3.14159265358979323846f;
  float w = (k == 0) ? rsqrtf((float)DN) : sqrtf(2.0f / (float)DN);
  float v = w * cosf(PI * (2.0f * (float)n + 1.0f) * (float)k / (2.0f * (float)DN));
  unsigned short hu = bf16_trunc(v);
  unsigned short lu = bf16_trunc(v - bf16_to_f(hu));
  Ah[n * DN + k] = hu;
  Al[n * DN + k] = lu;
}

// Split 16 f32 (two contiguous runs of 8) into hi/lo bf16 fragments,
// packed pairwise into dwords (maps to v_perm/v_pack, no b16 inserts).
__device__ __forceinline__ void split_runs(const float* __restrict__ r0,
                                           const float* __restrict__ r1,
                                           v16bf& hi, v16bf& lo) {
  v8i hw, lw;
#pragma unroll
  for (int p = 0; p < 8; ++p) {
    const float* rp = (p < 4) ? r0 : r1;
    const int q = (p & 3) * 2;
    float f0 = rp[q], f1 = rp[q + 1];
    unsigned int b0 = __float_as_uint(f0), b1 = __float_as_uint(f1);
    unsigned int h0 = b0 & 0xFFFF0000u,   h1 = b1 & 0xFFFF0000u;
    hw[p] = (b0 >> 16) | h1;
    float l0 = f0 - __uint_as_float(h0);
    float l1 = f1 - __uint_as_float(h1);
    lw[p] = (__float_as_uint(l0) >> 16) | (__float_as_uint(l1) & 0xFFFF0000u);
  }
  hi = __builtin_bit_cast(v16bf, hw);
  lo = __builtin_bit_cast(v16bf, lw);
}

__device__ __forceinline__ v8f wmma3(v16bf ah, v16bf al, v16bf bh, v16bf bl, v8f acc) {
  acc = __builtin_amdgcn_wmma_f32_16x16x32_bf16(false, al, false, bh, (short)0, acc, false, false);
  acc = __builtin_amdgcn_wmma_f32_16x16x32_bf16(false, ah, false, bl, (short)0, acc, false, false);
  acc = __builtin_amdgcn_wmma_f32_16x16x32_bf16(false, ah, false, bh, (short)0, acc, false, false);
  return acc;
}

__global__ void __launch_bounds__(256, 1)
idct2_wmma(const float* __restrict__ x, float* __restrict__ y,
           const unsigned short* __restrict__ Ah,
           const unsigned short* __restrict__ Al) {
  extern __shared__ float Tt[];  // Tt[n][m] = T[m][n], pitch 257 floats

  const int img  = blockIdx.x;
  const float* X = x + (size_t)img * DN * DN;
  float*       Y = y + (size_t)img * DN * DN;

  const int lane  = threadIdx.x & 31;
  const int wave  = threadIdx.x >> 5;   // 8 waves
  const int lrow  = lane & 15;
  const int lhalf = (lane >> 4) & 1;
  // wave tile block: 4 mt rows x 8 nt cols of 16x16 tiles
  const int mtb = (wave & 3) * 4;       // first mt tile of this wave
  const int ntb = (wave >> 2) * 8;      // first nt tile of this wave

  // -------- Stage 1: T = X * A^T (IDCT along last axis), store transposed ----
  for (int mi = 0; mi < 4; ++mi) {
    const int m0 = (mtb + mi) * 16;
    const float* xrow = X + (size_t)(m0 + lrow) * DN;
    for (int nb = 0; nb < 2; ++nb) {
      const int n0b = (ntb + nb * 4) * 16;
      v8f acc[4] = {{}, {}, {}, {}};
      for (int kc = 0; kc < 8; ++kc) {
        const int k0 = kc * 32;
        // A-frag (X tile): split once, reuse for 4 B tiles (12 WMMAs)
        const float* xr = xrow + k0 + lhalf * 16;
        v16bf xh, xl;
        split_runs(xr, xr + 8, xh, xl);
#pragma unroll
        for (int j = 0; j < 4; ++j) {
          const int n = n0b + j * 16 + lrow;
          const v16u bu  = *(const v16u*)(Ah + (size_t)n * DN + k0 + lhalf * 16);
          const v16u blu = *(const v16u*)(Al + (size_t)n * DN + k0 + lhalf * 16);
          acc[j] = wmma3(xh, xl,
                         __builtin_bit_cast(v16bf, bu),
                         __builtin_bit_cast(v16bf, blu), acc[j]);
        }
      }
#pragma unroll
      for (int j = 0; j < 4; ++j) {
        const int n = n0b + j * 16 + lrow;
#pragma unroll
        for (int r = 0; r < 8; ++r)
          Tt[(size_t)n * LDS_PITCH + (m0 + r + 8 * lhalf)] = acc[j][r];
      }
    }
  }

  __syncthreads();

  // -------- Stage 2: Y = A * T (IDCT along rows); B[k][n] = Tt[n][k] ---------
  for (int nj = 0; nj < 8; ++nj) {
    const int n0 = (ntb + nj) * 16;
    v8f acc[4] = {{}, {}, {}, {}};
    for (int kc = 0; kc < 8; ++kc) {
      const int k0 = kc * 32;
      // B-frag (T from LDS): split once, reuse for 4 A tiles (12 WMMAs)
      const float* tr = &Tt[(size_t)(n0 + lrow) * LDS_PITCH + k0 + lhalf * 8];
      v16bf th, tl;
      split_runs(tr, tr + 16, th, tl);
#pragma unroll
      for (int mi = 0; mi < 4; ++mi) {
        const int m = (mtb + mi) * 16 + lrow;
        const unsigned short* ar  = Ah + (size_t)m * DN + k0 + lhalf * 8;
        const unsigned short* arl = Al + (size_t)m * DN + k0 + lhalf * 8;
        const v8u a0 = *(const v8u*)ar,  a1 = *(const v8u*)(ar + 16);
        const v8u b0 = *(const v8u*)arl, b1 = *(const v8u*)(arl + 16);
        const v16u au  = __builtin_shufflevector(a0, a1, 0,1,2,3,4,5,6,7,8,9,10,11,12,13,14,15);
        const v16u alu = __builtin_shufflevector(b0, b1, 0,1,2,3,4,5,6,7,8,9,10,11,12,13,14,15);
        acc[mi] = wmma3(__builtin_bit_cast(v16bf, au),
                        __builtin_bit_cast(v16bf, alu), th, tl, acc[mi]);
      }
    }
#pragma unroll
    for (int mi = 0; mi < 4; ++mi) {
      const int m0 = (mtb + mi) * 16;
#pragma unroll
      for (int r = 0; r < 8; ++r)
        Y[(size_t)(m0 + r + 8 * lhalf) * DN + (n0 + lrow)] = acc[mi][r];
    }
  }
}

extern "C" void kernel_launch(void* const* d_in, const int* in_sizes, int n_in,
                              void* d_out, int out_size, void* d_ws, size_t ws_size,
                              hipStream_t stream) {
  const float* x = (const float*)d_in[0];
  float*       y = (float*)d_out;

  // workspace: Ah (64K bf16) | Al (64K bf16) = 256 KB
  unsigned short* Ah = (unsigned short*)d_ws;
  unsigned short* Al = Ah + DN * DN;

  build_basis<<<dim3(DN), dim3(DN), 0, stream>>>(Ah, Al);

  const int images = 16 * 32;                          // 512 images
  const size_t lds_bytes = (size_t)DN * LDS_PITCH * 4; // ~257 KB < 320 KB/WGP
  idct2_wmma<<<dim3(images), dim3(256), lds_bytes, stream>>>(x, y, Ah, Al);
}
